// ObjectClassifier_74363063763324
// MI455X (gfx1250) — matrix operand order repaired and checked by reading, not compiled, not run on hardware
//
#include <hip/hip_runtime.h>
#include <hip/hip_bf16.h>

typedef _Float16 v16h __attribute__((ext_vector_type(16)));
typedef _Float16 v8h  __attribute__((ext_vector_type(8)));
typedef float    v8f  __attribute__((ext_vector_type(8)));

#define N_ROWS   50000
#define NCLS     37
#define EMB      200
#define OBJ_DIM  2048
#define POSD     128
#define D_IN     2376      // OBJ_DIM + EMB + POSD
#define HID      1024
#define XT_W     352       // (EMB + POSD)=328 padded to 32-multiple
#define KPAD     2400      // OBJ_DIM + XT_W
#define BN_EPS   1e-5f
#define OUT_DIST (N_ROWS * NCLS)

// ---- workspace layout (bytes, all 256-aligned) ----
#define OFF_POS   0UL          // 8 f32: center_size sum[4], sumsq[4]
#define OFF_YSUM  256UL        // 1024 f32
#define OFF_YSSQ  4352UL       // 1024 f32
#define OFF_YSCL  8448UL       // 1024 f32: rstd*gamma
#define OFF_YSHF  12544UL      // 1024 f32: beta - mu*rstd*gamma
#define OFF_WT    16640UL      // int_W transposed f16: [1024][2400]
#define OFF_WD    4931840UL    // dec_W transposed f16: [48][1024]
#define OFF_XT    5030144UL    // f16 tail of A: [N_ROWS][352]
#define OFF_Y     40230144UL   // y column-major f16: [1024][N_ROWS]
#define WS_NEEDED (OFF_Y + (size_t)HID * N_ROWS * 2)   // ~143 MB

// WMMA helper
__device__ __forceinline__ v8f wmma_f16(v16h a, v16h b, v8f c) {
    return __builtin_amdgcn_wmma_f32_16x16x32_f16(false, a, false, b, (short)0, c,
                                                  false, false);
}

// One 32-deep k-slab: 2 B-frag global b128 pairs + 4 A-frag ds b128 pairs -> 8 WMMAs
__device__ __forceinline__ void tile_mma(const _Float16* Alds, const _Float16* __restrict__ WtTile,
                                         int wm, int wn, int lr, int half, v8f (&acc)[4][2]) {
    v16h bf[2];
#pragma unroll
    for (int ni = 0; ni < 2; ++ni) {             // B frag: column n, contiguous K run
        const _Float16* p = WtTile + (long)(wn * 32 + ni * 16 + lr) * KPAD + half * 16;
        union { v16h v; v8h h[2]; } u;
        u.h[0] = *(const v8h*)p;
        u.h[1] = *(const v8h*)(p + 8);
        bf[ni] = u.v;
    }
#pragma unroll
    for (int mi = 0; mi < 4; ++mi) {             // A frag: lanes L / L+16 share row, split K halves
        const _Float16* pa = &Alds[(wm * 64 + mi * 16 + lr) * 40 + half * 16];
        union { v16h v; v8h h[2]; } ua;
        ua.h[0] = *(const v8h*)pa;
        ua.h[1] = *(const v8h*)(pa + 8);
#pragma unroll
        for (int ni = 0; ni < 2; ++ni)
            acc[mi][ni] = wmma_f16(ua.v, bf[ni], acc[mi][ni]);
    }
}

// ---------------- K0: zero the atomic-accumulated stats ----------------
__global__ void k_init(char* ws) {
    int i = blockIdx.x * blockDim.x + threadIdx.x;
    float* pos = (float*)(ws + OFF_POS);
    float* ys  = (float*)(ws + OFF_YSUM);
    float* yq  = (float*)(ws + OFF_YSSQ);
    if (i < 8)    pos[i] = 0.f;
    if (i < HID) { ys[i] = 0.f; yq[i] = 0.f; }
}

// ---------------- K1: center_size BN statistics (sum / sumsq over N) ----------------
__global__ void k_pos_stats(const float* __restrict__ boxes, float* __restrict__ posStat) {
    float s[4] = {0.f, 0.f, 0.f, 0.f}, sq[4] = {0.f, 0.f, 0.f, 0.f};
    const int stride = gridDim.x * blockDim.x;
    for (long i = blockIdx.x * blockDim.x + threadIdx.x; i < N_ROWS; i += stride) {
        float b0 = boxes[i * 5 + 1], b1 = boxes[i * 5 + 2];
        float b2 = boxes[i * 5 + 3], b3 = boxes[i * 5 + 4];
        float c0 = (b0 + b2) * 0.5f, c1 = (b1 + b3) * 0.5f;
        float c2 = b2 - b0 + 1.f,    c3 = b3 - b1 + 1.f;
        s[0] += c0; sq[0] += c0 * c0;  s[1] += c1; sq[1] += c1 * c1;
        s[2] += c2; sq[2] += c2 * c2;  s[3] += c3; sq[3] += c3 * c3;
    }
#pragma unroll
    for (int j = 0; j < 4; ++j)
#pragma unroll
        for (int off = 16; off >= 1; off >>= 1) {
            s[j]  += __shfl_xor(s[j],  off);
            sq[j] += __shfl_xor(sq[j], off);
        }
    __shared__ float part[8][8];
    const int lane = threadIdx.x & 31, w = threadIdx.x >> 5;
    if (lane == 0) {
#pragma unroll
        for (int j = 0; j < 4; ++j) { part[w][j] = s[j]; part[w][4 + j] = sq[j]; }
    }
    __syncthreads();
    if (threadIdx.x < 8) {
        float v = 0.f;
#pragma unroll
        for (int ww = 0; ww < 8; ++ww) v += part[ww][threadIdx.x];
        atomicAdd(&posStat[threadIdx.x], v);
    }
}

// ---------------- K2: int_W (D_IN x HID f32) -> Wt (HID x KPAD f16, transposed, zero-pad) ----
__global__ void k_prep_wt(const float* __restrict__ intW, _Float16* __restrict__ Wt) {
    __shared__ float tile[32][33];
    const int k0 = blockIdx.x * 32;            // gridDim.x = 75
    const int h0 = blockIdx.y * 32;            // gridDim.y = 32
    const int tx = threadIdx.x, ty = threadIdx.y;   // (32, 8)
#pragma unroll
    for (int i = 0; i < 32; i += 8) {
        int k = k0 + ty + i;
        tile[ty + i][tx] = (k < D_IN) ? intW[(long)k * HID + h0 + tx] : 0.f;
    }
    __syncthreads();
#pragma unroll
    for (int i = 0; i < 32; i += 8) {
        int h = h0 + ty + i;
        Wt[(long)h * KPAD + k0 + tx] = (_Float16)tile[tx][ty + i];
    }
}

// ---------------- K3: dec_W (HID x 37 f32) -> Wd (48 x HID f16, transposed, zero-pad) ----
__global__ void k_prep_wd(const float* __restrict__ decW, _Float16* __restrict__ Wd) {
    int idx = blockIdx.x * blockDim.x + threadIdx.x;
    if (idx < 48 * HID) {
        int n = idx >> 10, k = idx & (HID - 1);
        Wd[idx] = (n < NCLS) ? (_Float16)decW[(long)k * NCLS + n] : (_Float16)0.f;
    }
}

// ---------------- K4: build f16 tail of A = [obj_embed | relu(pos) | 0-pad] ----------------
__global__ void k_build_xt(const float* __restrict__ dist, const float* __restrict__ boxes,
                           const float* __restrict__ objW, const float* __restrict__ posG,
                           const float* __restrict__ posB, const float* __restrict__ posW,
                           const float* __restrict__ posBias, const float* __restrict__ posStat,
                           _Float16* __restrict__ Xt) {
    __shared__ float sdist[36];
    __shared__ float spos[4];
    const int row = blockIdx.x;
    const int t = threadIdx.x;
    if (t < 36) sdist[t] = dist[(long)row * 36 + t];
    if (t < 4) {
        float b0 = boxes[(long)row * 5 + 1], b1 = boxes[(long)row * 5 + 2];
        float b2 = boxes[(long)row * 5 + 3], b3 = boxes[(long)row * 5 + 4];
        float cs = (t == 0) ? (b0 + b2) * 0.5f
                 : (t == 1) ? (b1 + b3) * 0.5f
                 : (t == 2) ? (b2 - b0 + 1.f) : (b3 - b1 + 1.f);
        float mu   = posStat[t] * (1.f / N_ROWS);
        float var  = posStat[4 + t] * (1.f / N_ROWS) - mu * mu;
        float rstd = 1.f / sqrtf(var + BN_EPS);
        spos[t] = (cs - mu) * rstd * posG[t] + posB[t];
    }
    __syncthreads();
    if (t < EMB) {
        float s = 0.f;
#pragma unroll
        for (int k = 0; k < 36; ++k) s += sdist[k] * objW[k * EMB + t];
        Xt[(long)row * XT_W + t] = (_Float16)s;
    } else if (t < EMB + POSD) {
        int c = t - EMB;
        float s = posBias[c];
#pragma unroll
        for (int k = 0; k < 4; ++k) s += spos[k] * posW[k * POSD + c];
        Xt[(long)row * XT_W + t] = (_Float16)fmaxf(s, 0.f);
    } else if (t < XT_W) {
        Xt[(long)row * XT_W + t] = (_Float16)0.f;
    }
}

// ---------------- K5: main GEMM  y[N][1024] = A[N][2400] x Wt^T, + column stats -------------
// block tile 128x128, 8 waves, wave tile 64x32, BK=32, y stored column-major f16.
// Branchless staging: OOB rows load a clamped address and are multiplied by 0 so padded
// rows produce y==0 exactly (zero contribution to the BN sum/sumsq atomics).
__global__ void __launch_bounds__(256) k_gemm_main(
        const float* __restrict__ features, const _Float16* __restrict__ Xt,
        const _Float16* __restrict__ Wt, _Float16* __restrict__ Ycm,
        float* __restrict__ ysum, float* __restrict__ yssq) {
    __shared__ _Float16 Alds[128 * 40];          // stride 40 halves -> conflict-free b128 frags
    const int tid  = threadIdx.x;
    const int lane = tid & 31, w = tid >> 5;
    const int wm = w >> 2, wn = w & 3;           // wave grid 2 x 4
    const int half = lane >> 4, lr = lane & 15;
    const int rowBase = blockIdx.y * 128;
    const int colBase = blockIdx.x * 128;

    v8f acc[4][2];
    const v8f vzero = {0.f, 0.f, 0.f, 0.f, 0.f, 0.f, 0.f, 0.f};
#pragma unroll
    for (int mi = 0; mi < 4; ++mi)
#pragma unroll
        for (int ni = 0; ni < 2; ++ni) acc[mi][ni] = vzero;

    const int  sr = tid & 127;                   // staging: row in tile
    const int  sc = (tid >> 7) * 16;             // staging: col 0 or 16
    const long grow  = rowBase + sr;
    const long growC = (grow < N_ROWS) ? grow : (N_ROWS - 1);   // clamped, always valid
    const float rscale = (grow < N_ROWS) ? 1.f : 0.f;           // zero out pad rows
    _Float16* const aldsDst = &Alds[sr * 40 + sc];              // 16B-aligned (80B row stride)
    const _Float16* const wtBase = Wt + (long)colBase * KPAD;

    // ---- phase 1: k in [0, 2048) from f32 features ----
    const float4* fsrc = (const float4*)(features + growC * (long)OBJ_DIM + sc);
    for (int kt = 0; kt < OBJ_DIM / 32; ++kt) {
        __syncthreads();
        v8h h0, h1;
#pragma unroll
        for (int q = 0; q < 2; ++q) {
            float4 f = fsrc[q];
            h0[q * 4 + 0] = (_Float16)(f.x * rscale); h0[q * 4 + 1] = (_Float16)(f.y * rscale);
            h0[q * 4 + 2] = (_Float16)(f.z * rscale); h0[q * 4 + 3] = (_Float16)(f.w * rscale);
        }
#pragma unroll
        for (int q = 0; q < 2; ++q) {
            float4 f = fsrc[2 + q];
            h1[q * 4 + 0] = (_Float16)(f.x * rscale); h1[q * 4 + 1] = (_Float16)(f.y * rscale);
            h1[q * 4 + 2] = (_Float16)(f.z * rscale); h1[q * 4 + 3] = (_Float16)(f.w * rscale);
        }
        *(v8h*)aldsDst       = h0;               // 2 x ds_store_b128
        *(v8h*)(aldsDst + 8) = h1;
        fsrc += 8;                               // advance 32 floats
        __syncthreads();
        tile_mma(Alds, wtBase + kt * 32, wm, wn, lr, half, acc);
    }

    // ---- phase 2: k in [2048, 2400) from prebuilt f16 tail ----
    const _Float16 hscale = (_Float16)rscale;
    const v8h* xsrc = (const v8h*)(Xt + growC * (long)XT_W + sc);
    for (int kt = 0; kt < XT_W / 32; ++kt) {
        __syncthreads();
        v8h a0 = xsrc[0], a1 = xsrc[1];
#pragma unroll
        for (int q = 0; q < 8; ++q) { a0[q] *= hscale; a1[q] *= hscale; }
        *(v8h*)aldsDst       = a0;
        *(v8h*)(aldsDst + 8) = a1;
        xsrc += 4;                               // advance 32 halves
        __syncthreads();
        tile_mma(Alds, wtBase + (OBJ_DIM + kt * 32), wm, wn, lr, half, acc);
    }

    // ---- per-column sum / sumsq -> global atomics (bias int_b cancels in BN, omitted) ----
#pragma unroll
    for (int ni = 0; ni < 2; ++ni) {
        float s = 0.f, sq = 0.f;
#pragma unroll
        for (int mi = 0; mi < 4; ++mi)
#pragma unroll
            for (int r = 0; r < 8; ++r) { float v = acc[mi][ni][r]; s += v; sq += v * v; }
        s  += __shfl_xor(s, 16);                 // combine complementary row halves (same column)
        sq += __shfl_xor(sq, 16);
        if (half == ni) {
            int col = colBase + wn * 32 + ni * 16 + lr;
            atomicAdd(&ysum[col], s);
            atomicAdd(&yssq[col], sq);
        }
    }

    // ---- store y column-major f16; frag row starts are multiples of 8 and N%8==0 ----
#pragma unroll
    for (int mi = 0; mi < 4; ++mi)
#pragma unroll
        for (int ni = 0; ni < 2; ++ni) {
            int col = colBase + wn * 32 + ni * 16 + lr;
            long r0 = rowBase + wm * 64 + mi * 16 + half * 8;
            if (r0 + 8 <= N_ROWS) {
                v8h o;
#pragma unroll
                for (int r = 0; r < 8; ++r) o[r] = (_Float16)acc[mi][ni][r];
                *(v8h*)(Ycm + (long)col * N_ROWS + r0) = o;
            }
        }
}

// ---------------- K6: finalize BN stats (fold into scale/shift) + pass-through labels -------
__global__ void k_finalize(const float* __restrict__ ysum, const float* __restrict__ yssq,
                           const float* __restrict__ gamma, const float* __restrict__ beta,
                           float* __restrict__ yscl, float* __restrict__ yshf,
                           const int* __restrict__ labels, int* __restrict__ outLabels) {
    int i = blockIdx.x * blockDim.x + threadIdx.x;
    if (i < HID) {
        float mu   = ysum[i] * (1.f / N_ROWS);
        float var  = yssq[i] * (1.f / N_ROWS) - mu * mu;
        float rstd = 1.f / sqrtf(var + BN_EPS);
        float a    = rstd * gamma[i];
        yscl[i] = a;                  // h = relu(y * a + b)
        yshf[i] = beta[i] - mu * a;
    }
    if (i < N_ROWS) outLabels[i] = labels[i];
}

// ---------------- K7: fused BN + ReLU + decoder GEMM (128 rows/block, 48-padded N) ----------
// Pad rows only affect guarded output stores (not statistics), so a pure address clamp
// suffices -- no zeroing, no exec divergence in the hot loop.
__global__ void __launch_bounds__(256) k_gemm_dec(
        const _Float16* __restrict__ Ycm, const _Float16* __restrict__ Wd,
        const float* __restrict__ yscl, const float* __restrict__ yshf,
        const float* __restrict__ decB, float* __restrict__ out) {
    __shared__ _Float16 Alds[128 * 40];
    const int tid  = threadIdx.x;
    const int lane = tid & 31, w = tid >> 5;     // wave w owns rows w*16 .. w*16+15
    const int half = lane >> 4, lr = lane & 15;
    const int rowBase = blockIdx.x * 128;

    v8f acc[3];
    const v8f vzero = {0.f, 0.f, 0.f, 0.f, 0.f, 0.f, 0.f, 0.f};
#pragma unroll
    for (int ni = 0; ni < 3; ++ni) acc[ni] = vzero;

    const int  sr = tid & 127;
    const int  sc = (tid >> 7) * 16;
    const long grow  = rowBase + sr;
    const long growC = (grow < N_ROWS) ? grow : (N_ROWS - 1);
    _Float16* const aldsDst = &Alds[sr * 40 + sc];

    for (int kt = 0; kt < HID / 32; ++kt) {
        __syncthreads();
        v8h h0, h1;
#pragma unroll
        for (int q = 0; q < 16; ++q) {
            int c = kt * 32 + sc + q;
            float v  = (float)Ycm[(long)c * N_ROWS + growC];
            float hh = fmaxf(0.f, fmaf(v, yscl[c], yshf[c]));
            if (q < 8) h0[q] = (_Float16)hh; else h1[q - 8] = (_Float16)hh;
        }
        *(v8h*)aldsDst       = h0;
        *(v8h*)(aldsDst + 8) = h1;
        __syncthreads();

        const _Float16* pa = &Alds[(w * 16 + lr) * 40 + half * 16];
        union { v16h v; v8h h[2]; } ua;
        ua.h[0] = *(const v8h*)pa;
        ua.h[1] = *(const v8h*)(pa + 8);
#pragma unroll
        for (int ni = 0; ni < 3; ++ni) {
            const _Float16* p = Wd + (long)(ni * 16 + lr) * HID + kt * 32 + half * 16;
            union { v16h v; v8h h[2]; } ub;
            ub.h[0] = *(const v8h*)p;
            ub.h[1] = *(const v8h*)(p + 8);
            acc[ni] = wmma_f16(ua.v, ub.v, acc[ni]);
        }
    }

#pragma unroll
    for (int ni = 0; ni < 3; ++ni) {
        int col = ni * 16 + lr;
        if (col < NCLS) {
            float bb = decB[col];
#pragma unroll
            for (int r = 0; r < 8; ++r) {
                long row = rowBase + w * 16 + half * 8 + r;
                if (row < N_ROWS) out[row * NCLS + col] = acc[ni][r] + bb;
            }
        }
    }
}

extern "C" void kernel_launch(void* const* d_in, const int* in_sizes, int n_in,
                              void* d_out, int out_size, void* d_ws, size_t ws_size,
                              hipStream_t stream) {
    const float* distribution = (const float*)d_in[0];
    const float* boxes        = (const float*)d_in[1];
    const float* features     = (const float*)d_in[2];
    const int*   labels       = (const int*)d_in[3];
    const float* objW         = (const float*)d_in[4];
    const float* posG         = (const float*)d_in[5];
    const float* posBeta      = (const float*)d_in[6];
    const float* posW         = (const float*)d_in[7];
    const float* posBias      = (const float*)d_in[8];
    const float* intW         = (const float*)d_in[9];
    // d_in[10] = int_b: unused — a pre-BN bias cancels exactly in BN-train
    const float* intG         = (const float*)d_in[11];
    const float* intBeta      = (const float*)d_in[12];
    const float* decW         = (const float*)d_in[13];
    const float* decB         = (const float*)d_in[14];
    (void)in_sizes; (void)n_in; (void)out_size;

    if (ws_size < WS_NEEDED) return;             // deterministic guard against undersized scratch

    char*      ws      = (char*)d_ws;
    float*     posStat = (float*)(ws + OFF_POS);
    float*     ysum    = (float*)(ws + OFF_YSUM);
    float*     yssq    = (float*)(ws + OFF_YSSQ);
    float*     yscl    = (float*)(ws + OFF_YSCL);
    float*     yshf    = (float*)(ws + OFF_YSHF);
    _Float16*  Wt      = (_Float16*)(ws + OFF_WT);
    _Float16*  Wd      = (_Float16*)(ws + OFF_WD);
    _Float16*  Xt      = (_Float16*)(ws + OFF_XT);
    _Float16*  Ycm     = (_Float16*)(ws + OFF_Y);
    float*     outF    = (float*)d_out;
    int*       outLab  = (int*)(outF + OUT_DIST);

    k_init     <<<4, 256, 0, stream>>>(ws);
    k_pos_stats<<<120, 256, 0, stream>>>(boxes, posStat);
    k_prep_wt  <<<dim3(KPAD / 32, HID / 32), dim3(32, 8), 0, stream>>>(intW, Wt);
    k_prep_wd  <<<(48 * HID + 255) / 256, 256, 0, stream>>>(decW, Wd);
    k_build_xt <<<N_ROWS, 384, 0, stream>>>(distribution, boxes, objW, posG, posBeta,
                                            posW, posBias, posStat, Xt);
    // grid.x = column tiles (8) fastest so concurrent blocks share the same A row-strip in L2
    k_gemm_main<<<dim3(HID / 128, (N_ROWS + 127) / 128), 256, 0, stream>>>(
        features, Xt, Wt, Ycm, ysum, yssq);
    k_finalize <<<(N_ROWS + 255) / 256, 256, 0, stream>>>(ysum, yssq, intG, intBeta,
                                                          yscl, yshf, labels, outLab);
    k_gemm_dec <<<(N_ROWS + 127) / 128, 256, 0, stream>>>(Ycm, Wd, yscl, yshf, decB, outF);
}
